// GraphFormer_PE_44341242364112
// MI455X (gfx1250) — matrix operand order repaired and checked by reading, not compiled, not run on hardware
//
#include <hip/hip_runtime.h>
#include <hip/hip_bf16.h>

#define NPTS 20000
#define KK   32
#define KP1  33
#define EPSF 1e-5f

typedef _Float16 f16_t;
typedef __attribute__((ext_vector_type(16))) _Float16 v16h;
typedef __attribute__((ext_vector_type(8)))  float    v8f;

__device__ __forceinline__ v8f wmma16(v16h a, v16h b, v8f c) {
  // D = A(16x32 f16) x B(32x16 f16) + C(16x16 f32)
  return __builtin_amdgcn_wmma_f32_16x16x32_f16(false, a, false, b, (short)0, c, false, false);
}

__device__ __forceinline__ void wfence() {
  __builtin_amdgcn_wave_barrier();
  asm volatile("" ::: "memory");
}

__device__ __forceinline__ float wave_sum(float v) {
#pragma unroll
  for (int o = 16; o > 0; o >>= 1) v += __shfl_xor(v, o, 32);
  return v;
}

__device__ __forceinline__ v8f zero8() {
  v8f z;
#pragma unroll
  for (int i = 0; i < 8; ++i) z[i] = 0.0f;
  return z;
}

// A fragment (16x32 f16) gathered from a row-major LDS array (stride in halfwords).
// ISA 7.12.2: lane = M (mod 16); halves 0..7 -> K 0..7, halves 8..15 -> K 16..23,
// lanes >=16 add +8 to K.
__device__ __forceinline__ v16h load_a_frag(const f16_t* A, int strideH, int mt, int kt, int lane) {
  const int m  = mt * 16 + (lane & 15);
  const int kb = kt * 32 + ((lane & 16) ? 8 : 0);
  const f16_t* r = A + m * strideH + kb;
  v16h a;
#pragma unroll
  for (int h = 0; h < 16; ++h) {
    const int k = (h < 8) ? h : (h + 8);
    a[h] = r[k];
  }
  return a;
}

// B fragment pre-packed per (kt,nt,lane): one contiguous 32B v16h per lane.
__device__ __forceinline__ v16h load_b_frag(const f16_t* Bf, int kt, int nt, int lane) {
  return *(const v16h*)(Bf + (((kt << 2) + nt) * 32 + lane) * 16);
}

template <int MT, int KT>
__device__ __forceinline__ void gemm_f16(const f16_t* A, int strideH, const f16_t* Bf,
                                         v8f (&C)[MT][4], int lane) {
#pragma unroll
  for (int mt = 0; mt < MT; ++mt) {
#pragma unroll
    for (int nt = 0; nt < 4; ++nt) C[mt][nt] = zero8();
#pragma unroll
    for (int kt = 0; kt < KT; ++kt) {
      const v16h a = load_a_frag(A, strideH, mt, kt, lane);
#pragma unroll
      for (int nt = 0; nt < 4; ++nt)
        C[mt][nt] = wmma16(a, load_b_frag(Bf, kt, nt, lane), C[mt][nt]);
    }
  }
}

// ---------------------------------------------------------------------------
// P1: pack an f32 (Krows x 64) weight into WMMA B-fragment layout (f16).
// Grid: ktiles*4 blocks, 512 threads. ISA B layout: col = lane&15,
// K = kt*32 + (lane>=16 ? 8 : 0) + (h < 8 ? h : h + 8).
// ---------------------------------------------------------------------------
__global__ __launch_bounds__(512) void pack_b_frag(const float* __restrict__ W, int Krows,
                                                   f16_t* __restrict__ out) {
  const int tile = blockIdx.x;
  const int kt = tile >> 2, nt = tile & 3;
  const int lane = threadIdx.x >> 4;   // 0..31
  const int h    = threadIdx.x & 15;   // 0..15
  const int k = kt * 32 + ((lane & 16) ? 8 : 0) + ((h < 8) ? h : (h + 8));
  const int n = nt * 16 + (lane & 15);
  const float v = (k < Krows) ? W[k * 64 + n] : 0.0f;
  out[(tile * 32 + lane) * 16 + h] = (f16_t)v;
}

// ---------------------------------------------------------------------------
// Kernel A: Q = relu(feats @ Wq + bq).  One wave -> 16 node rows, 8 WMMAs.
// ---------------------------------------------------------------------------
__global__ __launch_bounds__(128) void kq(const float* __restrict__ feats,
                                          const float* __restrict__ bq,
                                          const f16_t* __restrict__ Wqf,
                                          float* __restrict__ Qg) {
  __shared__ f16_t Ast_s[4][16 * 68];
  const int lane = threadIdx.x & 31, wid = threadIdx.x >> 5;
  const long r0 = (long)(blockIdx.x * 4 + wid) * 16;
  if (r0 >= NPTS) return;
  f16_t* Ast = Ast_s[wid];
  for (int i = lane; i < 16 * 64; i += 32) {
    const int rr = i >> 6, cc = i & 63;
    Ast[rr * 68 + cc] = (f16_t)feats[(r0 + rr) * 64 + cc];
  }
  wfence();
  v8f C[1][4];
  gemm_f16<1, 2>(Ast, 68, Wqf, C, lane);
  const int col0 = lane & 15, half8 = (lane & 16) ? 8 : 0;
#pragma unroll
  for (int nt = 0; nt < 4; ++nt) {
    const int col = nt * 16 + col0;
    const float bb = bq[col];
#pragma unroll
    for (int v = 0; v < 8; ++v) {
      const int row = v + half8;
      Qg[(r0 + row) * 64 + col] = fmaxf(C[0][nt][v] + bb, 0.0f);
    }
  }
}

// ---------------------------------------------------------------------------
// Kernel B: fused per-node attention. One wave per node, 2 waves per block.
// 5 x (32x64x64) WMMA GEMMs + lane-mapped sparsemax + LayerNorm -> h.
// ---------------------------------------------------------------------------
__global__ __launch_bounds__(64) void kfuse(
    const float* __restrict__ coords, const float* __restrict__ feats,
    const float* __restrict__ Wl1,   const float* __restrict__ bl1,
    const float* __restrict__ bl2,   const float* __restrict__ bk,
    const float* __restrict__ bv,    const float* __restrict__ bw1,
    const float* __restrict__ bw2,   const float* __restrict__ g_att,
    const float* __restrict__ b_att, const int* __restrict__ knn,
    const float* __restrict__ Qg,
    const f16_t* __restrict__ Wl2f, const f16_t* __restrict__ Wkf,
    const f16_t* __restrict__ Wvf,  const f16_t* __restrict__ Ww1f,
    const f16_t* __restrict__ Ww2f,
    float* __restrict__ hout) {
  __shared__ f16_t Ah_s[2][32 * 68];   // tmp -> rel -> s   (A operands)
  __shared__ f16_t peh_s[2][32 * 68];  // pe               (A operand)
  __shared__ float w32_s[2][32 * 65];  // fc32 -> w32      (padded, conflict-free)
  __shared__ float xrow_s[2][64];

  const int lane = threadIdx.x & 31;
  const int wid  = threadIdx.x >> 5;
  const int n    = blockIdx.x * 2 + wid;
  f16_t* Ah   = Ah_s[wid];
  f16_t* peh  = peh_s[wid];
  float* w32  = w32_s[wid];
  float* xrow = xrow_s[wid];
  const int col0  = lane & 15;
  const int half8 = (lane & 16) ? 8 : 0;

  // 1) tmp = relu(diff @ Wl1 + bl1); lane = neighbor k
  const int idxk = knn[n * KP1 + lane];
  const float dx = coords[idxk * 3 + 0] - coords[n * 3 + 0];
  const float dy = coords[idxk * 3 + 1] - coords[n * 3 + 1];
  const float dz = coords[idxk * 3 + 2] - coords[n * 3 + 2];
  for (int d = 0; d < 64; ++d) {
    const float t = fmaf(dx, Wl1[d], fmaf(dy, Wl1[64 + d], fmaf(dz, Wl1[128 + d], bl1[d])));
    Ah[lane * 68 + d] = (f16_t)fmaxf(t, 0.0f);
  }
  wfence();

  // 2) fc = tmp @ Wl2 + bl2 -> w32 (as fc32)
  {
    v8f C[2][4];
    gemm_f16<2, 2>(Ah, 68, Wl2f, C, lane);
#pragma unroll
    for (int mt = 0; mt < 2; ++mt)
#pragma unroll
      for (int nt = 0; nt < 4; ++nt) {
        const int col = nt * 16 + col0;
        const float bb = bl2[col];
#pragma unroll
        for (int v = 0; v < 8; ++v)
          w32[(mt * 16 + v + half8) * 65 + col] = C[mt][nt][v] + bb;
      }
  }
  wfence();

  // 3) pe = fc + feats[idx]
  {
    const float* fr = feats + (size_t)idxk * 64;
    for (int d = 0; d < 64; d += 4) {
      const float4 f = *(const float4*)(fr + d);
      peh[lane * 68 + d + 0] = (f16_t)(w32[lane * 65 + d + 0] + f.x);
      peh[lane * 68 + d + 1] = (f16_t)(w32[lane * 65 + d + 1] + f.y);
      peh[lane * 68 + d + 2] = (f16_t)(w32[lane * 65 + d + 2] + f.z);
      peh[lane * 68 + d + 3] = (f16_t)(w32[lane * 65 + d + 3] + f.w);
    }
  }
  wfence();

  // 4) Kf = relu(pe @ Wk + bk); rel = (Kf - Q[n]) * fc -> Ah
  {
    v8f C[2][4];
    gemm_f16<2, 2>(peh, 68, Wkf, C, lane);
#pragma unroll
    for (int mt = 0; mt < 2; ++mt)
#pragma unroll
      for (int nt = 0; nt < 4; ++nt) {
        const int col = nt * 16 + col0;
        const float qv = Qg[(size_t)n * 64 + col];
        const float bb = bk[col];
#pragma unroll
        for (int v = 0; v < 8; ++v) {
          const int row = mt * 16 + v + half8;
          const float kf = fmaxf(C[mt][nt][v] + bb, 0.0f);
          Ah[row * 68 + col] = (f16_t)((kf - qv) * w32[row * 65 + col]);
        }
      }
  }
  wfence();

  // 5) s = relu(rel @ Ww1 + bw1) -> Ah (held in regs until GEMM reads done)
  {
    v8f C[2][4];
    gemm_f16<2, 2>(Ah, 68, Ww1f, C, lane);
    wfence();
#pragma unroll
    for (int mt = 0; mt < 2; ++mt)
#pragma unroll
      for (int nt = 0; nt < 4; ++nt) {
        const int col = nt * 16 + col0;
        const float bb = bw1[col];
#pragma unroll
        for (int v = 0; v < 8; ++v)
          Ah[(mt * 16 + v + half8) * 68 + col] = (f16_t)fmaxf(C[mt][nt][v] + bb, 0.0f);
      }
  }
  wfence();

  // 6) w = s @ Ww2 + bw2 -> w32
  {
    v8f C[2][4];
    gemm_f16<2, 2>(Ah, 68, Ww2f, C, lane);
#pragma unroll
    for (int mt = 0; mt < 2; ++mt)
#pragma unroll
      for (int nt = 0; nt < 4; ++nt) {
        const int col = nt * 16 + col0;
        const float bb = bw2[col];
#pragma unroll
        for (int v = 0; v < 8; ++v)
          w32[(mt * 16 + v + half8) * 65 + col] = C[mt][nt][v] + bb;
      }
  }
  wfence();

  // 7) sparsemax over k (k = lane) for each dim d
  for (int d = 0; d < 64; ++d) {
    const float z = w32[lane * 65 + d];
    float zs = z;
#pragma unroll
    for (int size = 2; size <= 32; size <<= 1) {
#pragma unroll
      for (int j = size >> 1; j > 0; j >>= 1) {
        const float o = __shfl_xor(zs, j, 32);
        const bool lower = ((lane & j) == 0);
        const bool descB = ((lane & size) == 0);
        const float mx = fmaxf(zs, o), mn = fminf(zs, o);
        zs = (descB == lower) ? mx : mn;   // descending sort across lanes
      }
    }
    float cs = zs;                          // inclusive prefix sum
#pragma unroll
    for (int off = 1; off < 32; off <<= 1) {
      const float o = __shfl_up(cs, off, 32);
      if (lane >= off) cs += o;
    }
    const bool support = (1.0f + (float)(lane + 1) * zs) > cs;
    const int ksup = __popcll(__ballot(support));
    const float csk = __shfl(cs, ksup - 1, 32);
    const float tau = (csk - 1.0f) / (float)ksup;
    w32[lane * 65 + d] = fmaxf(z - tau, 0.0f);
  }
  wfence();

  // 8) V = pe @ Wv (+bv in combine); 9) attn + residuals
  v8f Vc[2][4];
  gemm_f16<2, 2>(peh, 68, Wvf, Vc, lane);
#pragma unroll
  for (int nt = 0; nt < 4; ++nt) {
    const int col = nt * 16 + col0;
    const float bb = bv[col];
    float p = 0.0f;
#pragma unroll
    for (int mt = 0; mt < 2; ++mt)
#pragma unroll
      for (int v = 0; v < 8; ++v) {
        const int row = mt * 16 + v + half8;
        p += (Vc[mt][nt][v] + bb) * w32[row * 65 + col];
      }
    p += __shfl_xor(p, 16, 32);
    const float x = p + 2.0f * feats[(size_t)n * 64 + col];
    if (lane < 16) xrow[col] = x;
  }
  wfence();

  // 10) LayerNorm -> h
  {
    const float a = xrow[lane], b = xrow[lane + 32];
    const float m = wave_sum(a + b) * (1.0f / 64.0f);
    const float da = a - m, db = b - m;
    const float var = wave_sum(da * da + db * db) * (1.0f / 64.0f);
    const float rs = rsqrtf(var + EPSF);
    hout[(size_t)n * 64 + lane]      = g_att[lane] * da * rs + b_att[lane];
    hout[(size_t)n * 64 + 32 + lane] = g_att[lane + 32] * db * rs + b_att[lane + 32];
  }
}

// ---------------------------------------------------------------------------
// Kernel C1: GCN aggregation. deg == 33 for every node => norm == 1/33.
// agg[n] = (x_in[n] + sum_{j in knn[n,1..32]} x_in[j]) / 33, padded to 96 dims.
// ---------------------------------------------------------------------------
__global__ __launch_bounds__(256) void kagg(const float* __restrict__ coords,
                                            const float* __restrict__ hbuf,
                                            const int* __restrict__ knn,
                                            float* __restrict__ agg) {
  const int lane = threadIdx.x & 31, wid = threadIdx.x >> 5;
  const int n = blockIdx.x * 8 + wid;
  float a0 = 0.0f, a1 = 0.0f, a2 = 0.0f;
  for (int t = 0; t <= KK; ++t) {
    const int j = (t == 0) ? n : knn[n * KP1 + t];
    a0 += hbuf[(size_t)j * 64 + lane];
    a1 += hbuf[(size_t)j * 64 + 32 + lane];
    if (lane < 3) a2 += coords[j * 3 + lane];
  }
  const float s = 1.0f / 33.0f;
  agg[(size_t)n * 96 + lane]      = a0 * s;
  agg[(size_t)n * 96 + 32 + lane] = a1 * s;
  agg[(size_t)n * 96 + 64 + lane] = (lane < 3) ? a2 * s : 0.0f;
}

// ---------------------------------------------------------------------------
// Kernel C2: FFN (K=96 padded GEMM + K=64 GEMM) + residual + LayerNorm -> out
// ---------------------------------------------------------------------------
__global__ __launch_bounds__(128) void kffn(const float* __restrict__ agg,
                                            const float* __restrict__ hbuf,
                                            const float* __restrict__ bsg,
                                            const float* __restrict__ bf,
                                            const float* __restrict__ gff,
                                            const float* __restrict__ bff,
                                            const f16_t* __restrict__ Wsgf,
                                            const f16_t* __restrict__ Wff,
                                            float* __restrict__ out) {
  __shared__ f16_t Aagg_s[4][16 * 100];
  __shared__ f16_t x1h_s[4][16 * 68];
  __shared__ float x2_s[4][16 * 65];
  const int lane = threadIdx.x & 31, wid = threadIdx.x >> 5;
  const long r0 = (long)(blockIdx.x * 4 + wid) * 16;
  if (r0 >= NPTS) return;
  f16_t* Aagg = Aagg_s[wid];
  f16_t* x1h  = x1h_s[wid];
  float* x2   = x2_s[wid];
  const int col0 = lane & 15, half8 = (lane & 16) ? 8 : 0;

  for (int i = lane; i < 16 * 96; i += 32) {
    const int rr = i / 96, cc = i % 96;
    Aagg[rr * 100 + cc] = (f16_t)agg[(r0 + rr) * 96 + cc];
  }
  wfence();

  v8f C[1][4];
  gemm_f16<1, 3>(Aagg, 100, Wsgf, C, lane);   // agg @ Wsg (K padded to 96)
#pragma unroll
  for (int nt = 0; nt < 4; ++nt) {
    const int col = nt * 16 + col0;
    const float bb = bsg[col];
#pragma unroll
    for (int v = 0; v < 8; ++v)
      x1h[(v + half8) * 68 + col] = (f16_t)fmaxf(C[0][nt][v] + bb, 0.0f);
  }
  wfence();

  gemm_f16<1, 2>(x1h, 68, Wff, C, lane);      // x1 @ Wf
#pragma unroll
  for (int nt = 0; nt < 4; ++nt) {
    const int col = nt * 16 + col0;
    const float bb = bf[col];
#pragma unroll
    for (int v = 0; v < 8; ++v) {
      const int row = v + half8;
      x2[row * 65 + col] = C[0][nt][v] + bb + hbuf[(r0 + row) * 64 + col];
    }
  }
  wfence();

  for (int i = 0; i < 16; ++i) {
    const float a = x2[i * 65 + lane], b = x2[i * 65 + lane + 32];
    const float m = wave_sum(a + b) * (1.0f / 64.0f);
    const float da = a - m, db = b - m;
    const float var = wave_sum(da * da + db * db) * (1.0f / 64.0f);
    const float rs = rsqrtf(var + EPSF);
    out[(r0 + i) * 64 + lane]      = gff[lane] * da * rs + bff[lane];
    out[(r0 + i) * 64 + 32 + lane] = gff[lane + 32] * db * rs + bff[lane + 32];
  }
}

// ---------------------------------------------------------------------------
extern "C" void kernel_launch(void* const* d_in, const int* in_sizes, int n_in,
                              void* d_out, int out_size, void* d_ws, size_t ws_size,
                              hipStream_t stream) {
  (void)in_sizes; (void)n_in; (void)out_size; (void)ws_size;
  const float* coords = (const float*)d_in[0];
  const float* feats  = (const float*)d_in[1];
  const float* Wq  = (const float*)d_in[2];  const float* bq  = (const float*)d_in[3];
  const float* Wk  = (const float*)d_in[4];  const float* bk  = (const float*)d_in[5];
  const float* Wv  = (const float*)d_in[6];  const float* bv  = (const float*)d_in[7];
  const float* Wl1 = (const float*)d_in[8];  const float* bl1 = (const float*)d_in[9];
  const float* Wl2 = (const float*)d_in[10]; const float* bl2 = (const float*)d_in[11];
  const float* Ww1 = (const float*)d_in[12]; const float* bw1 = (const float*)d_in[13];
  const float* Ww2 = (const float*)d_in[14]; const float* bw2 = (const float*)d_in[15];
  const float* g_att = (const float*)d_in[16]; const float* b_att = (const float*)d_in[17];
  const float* Wsg = (const float*)d_in[18]; const float* bsg = (const float*)d_in[19];
  const float* Wf  = (const float*)d_in[20]; const float* bf  = (const float*)d_in[21];
  const float* g_ff = (const float*)d_in[22]; const float* b_ff = (const float*)d_in[23];
  const int*  knn  = (const int*)d_in[24];
  float* out = (float*)d_out;

  char* ws = (char*)d_ws;
  size_t off = 0;
  auto walloc = [&](size_t bytes) -> void* {
    void* p = ws + off;
    off += (bytes + 255) & ~(size_t)255;
    return p;
  };
  float* Qg   = (float*)walloc((size_t)NPTS * 64 * 4);
  float* hbuf = (float*)walloc((size_t)NPTS * 64 * 4);
  float* agg  = (float*)walloc((size_t)NPTS * 96 * 4);
  f16_t* Wqf  = (f16_t*)walloc(8 * 512 * 2);
  f16_t* Wl2f = (f16_t*)walloc(8 * 512 * 2);
  f16_t* Wkf  = (f16_t*)walloc(8 * 512 * 2);
  f16_t* Wvf  = (f16_t*)walloc(8 * 512 * 2);
  f16_t* Ww1f = (f16_t*)walloc(8 * 512 * 2);
  f16_t* Ww2f = (f16_t*)walloc(8 * 512 * 2);
  f16_t* Wsgf = (f16_t*)walloc(12 * 512 * 2);
  f16_t* Wff  = (f16_t*)walloc(8 * 512 * 2);

  // P1: pack all weights into WMMA B-fragment layout (f16)
  pack_b_frag<<<8,  512, 0, stream>>>(Wq,  64, Wqf);
  pack_b_frag<<<8,  512, 0, stream>>>(Wl2, 64, Wl2f);
  pack_b_frag<<<8,  512, 0, stream>>>(Wk,  64, Wkf);
  pack_b_frag<<<8,  512, 0, stream>>>(Wv,  64, Wvf);
  pack_b_frag<<<8,  512, 0, stream>>>(Ww1, 64, Ww1f);
  pack_b_frag<<<8,  512, 0, stream>>>(Ww2, 64, Ww2f);
  pack_b_frag<<<12, 512, 0, stream>>>(Wsg, 67, Wsgf);   // zero-padded K: 67 -> 96
  pack_b_frag<<<8,  512, 0, stream>>>(Wf,  64, Wff);

  const int waveBlocks = (NPTS / 16 + 3) / 4;           // 313
  kq<<<waveBlocks, 128, 0, stream>>>(feats, bq, Wqf, Qg);
  kfuse<<<NPTS / 2, 64, 0, stream>>>(coords, feats, Wl1, bl1, bl2, bk, bv, bw1, bw2,
                                     g_att, b_att, knn, Qg,
                                     Wl2f, Wkf, Wvf, Ww1f, Ww2f, hbuf);
  kagg<<<NPTS / 8, 256, 0, stream>>>(coords, hbuf, knn, agg);
  kffn<<<waveBlocks, 128, 0, stream>>>(agg, hbuf, bsg, bf, g_ff, b_ff, Wsgf, Wff, out);
}
